// RelationalRegulatoryGNN_78769700208706
// MI455X (gfx1250) — matrix operand order repaired and checked by reading, not compiled, not run on hardware
//
#include <hip/hip_runtime.h>
#include <math.h>

// Problem constants (match reference)
#define HB   8         // batch
#define NN   10000     // nodes
#define HH   64        // hidden
#define LL   4         // layers
#define RR   3         // relations
#define EE   200000    // edges per relation
#define FPD  2048      // fingerprint dim
#define MROWS (HB * NN)        // 80000 rows
#define STRIPS (MROWS / 16)    // 5000 strips of 16 rows
#define SPW  5                 // strips per wave in GEMM

typedef __attribute__((ext_vector_type(16))) _Float16 v16h;
typedef __attribute__((ext_vector_type(8)))  _Float16 v8h;
typedef __attribute__((ext_vector_type(2)))  _Float16 v2h;
typedef __attribute__((ext_vector_type(8)))  float    v8f;

// ---------------------------------------------------------------------------
// FiLM MLP: per batch b, h1 = relu(fp[b] @ W_f1 + b_f1);
// film = h1 @ W_f2 + b_f2. Store [8][128]: [0:64)=(1+tanh(gamma)), [64:128)=beta
// ---------------------------------------------------------------------------
__global__ void film_kernel(const float* __restrict__ fp,
                            const float* __restrict__ Wf1,
                            const float* __restrict__ bf1,
                            const float* __restrict__ Wf2,
                            const float* __restrict__ bf2,
                            float* __restrict__ film) {
  __shared__ float h1[HH];
  const int b = blockIdx.x;
  const int t = threadIdx.x;   // 128 threads
  if (t < HH) {
    float s = bf1[t];
    const float* fpb = fp + (size_t)b * FPD;
    for (int k = 0; k < FPD; ++k) s += fpb[k] * Wf1[k * HH + t];
    h1[t] = fmaxf(s, 0.0f);
  }
  __syncthreads();
  float s = bf2[t];
  for (int k = 0; k < HH; ++k) s += h1[k] * Wf2[k * (2 * HH) + t];
  film[b * 128 + t] = (t < HH) ? (1.0f + tanhf(s)) : s;
}

// ---------------------------------------------------------------------------
// xh0 = f16( relu(ctl*Wse+bse) + relu(tg*Wse+bse) + cell_emb[cidx[b]], FiLM )
// ---------------------------------------------------------------------------
__global__ void init_x(const float* __restrict__ ctl,
                       const float* __restrict__ tg,
                       const float* __restrict__ Wse,
                       const float* __restrict__ bse,
                       const float* __restrict__ cemb,
                       const int*   __restrict__ cidx,
                       const float* __restrict__ film,
                       _Float16* __restrict__ Xh) {
  size_t i = (size_t)blockIdx.x * blockDim.x + threadIdx.x;  // over 5,120,000
  int hcol = (int)(i & 63);
  size_t row = i >> 6;                 // b*NN + n
  int b = (int)(row / NN);
  float c = ctl[row], d = tg[row];
  float w = Wse[hcol], bb = bse[hcol];
  float v = fmaxf(fmaf(c, w, bb), 0.0f) + fmaxf(fmaf(d, w, bb), 0.0f)
          + cemb[(size_t)cidx[b] * HH + hcol];
  v = fmaf(v, film[b * 128 + hcol], film[b * 128 + HH + hcol]);
  Xh[i] = (_Float16)v;
}

// ---------------------------------------------------------------------------
// Y[M,64] = Xh[M,64] @ W[64,64] via v_wmma_f32_16x16x32_f16 (f32 accumulate).
// Weight staged in LDS pre-swizzled into B-fragment layout:
//   Wl[((c*4+t)*32 + lane)*16 + j] = W[c*32 + (lane>>4)*16 + j][t*16 + (lane&15)]
// so each fragment is one contiguous 32B LDS read. A fragments are two
// contiguous 16B runs from the f16 activation mirror. Each wave caches all 8
// B fragments in registers and sweeps SPW row-strips (40 WMMAs / wave).
// ---------------------------------------------------------------------------
__global__ __launch_bounds__(256) void gemm64(const _Float16* __restrict__ Xh,
                                              const float* __restrict__ W,
                                              float* __restrict__ Y) {
  __shared__ _Float16 Wl[2 * 4 * 32 * 16];   // 8 KB
  const int tid = threadIdx.x;
  {
    const int nn = tid & 15, t = (tid >> 4) & 3, h2 = (tid >> 6) & 1, c = tid >> 7;
    const int n = t * 16 + nn, lane0 = h2 * 16 + nn;
    v16h tmp;
#pragma unroll
    for (int j = 0; j < 16; ++j)
      tmp[j] = (_Float16)W[(c * 32 + h2 * 16 + j) * HH + n];
    *(v16h*)&Wl[((c * 4 + t) * 32 + lane0) * 16] = tmp;
  }
  __syncthreads();

  const int wave = tid >> 5, lane = tid & 31;
  const int m = lane & 15, hi = lane >> 4;
  const int ksel = hi * 16;

  // Cache all 8 B fragments (2 k-chunks x 4 n-tiles) in registers.
  v16h bf[2][4];
#pragma unroll
  for (int c = 0; c < 2; ++c)
#pragma unroll
    for (int t = 0; t < 4; ++t)
      bf[c][t] = *(const v16h*)&Wl[((c * 4 + t) * 32 + lane) * 16];

  const int strip0 = (blockIdx.x * 8 + wave) * SPW;   // 125 blocks * 8 * 5 = 5000
#pragma unroll
  for (int s = 0; s < SPW; ++s) {
    const int strip = strip0 + s;
    const _Float16* xr = Xh + (size_t)(strip * 16 + m) * HH;
    v16h a[2];
#pragma unroll
    for (int c = 0; c < 2; ++c) {
      v8h lo = *(const v8h*)(xr + c * 32 + ksel);
      v8h hi8 = *(const v8h*)(xr + c * 32 + ksel + 8);
#pragma unroll
      for (int j = 0; j < 8; ++j) { a[c][j] = lo[j]; a[c][8 + j] = hi8[j]; }
    }
#pragma unroll
    for (int t = 0; t < 4; ++t) {
      v8f acc = {};
      acc = __builtin_amdgcn_wmma_f32_16x16x32_f16(false, a[0], false, bf[0][t],
                                                   (short)0, acc, false, false);
      acc = __builtin_amdgcn_wmma_f32_16x16x32_f16(false, a[1], false, bf[1][t],
                                                   (short)0, acc, false, false);
      const int col = t * 16 + m;
#pragma unroll
      for (int i = 0; i < 8; ++i)
        Y[(size_t)(strip * 16 + hi * 8 + i) * HH + col] = acc[i];
    }
  }
}

// ---------------------------------------------------------------------------
// Scatter: one wave per edge; for all 8 batches: out[b,dst,:] += w * h[b,src,:]
// float4 gathers; h/out are 20.5 MB each -> atomics resolve in L2 (192 MB).
// ---------------------------------------------------------------------------
__global__ __launch_bounds__(256) void scatter_edges(const float* __restrict__ Hbuf,
                                                     const int*   __restrict__ src,
                                                     const int*   __restrict__ dst,
                                                     const float* __restrict__ ew,
                                                     float* __restrict__ Out) {
  const int e = (blockIdx.x * 256 + threadIdx.x) >> 5;
  const int lane = threadIdx.x & 31;
  if (e >= EE) return;
  const int s = src[e], d = dst[e];
  const float w = ew[e];
  const int c4 = lane & 15, bsel = lane >> 4;   // lane -> 4 cols, 2 batch phases
  const float4* hv = (const float4*)Hbuf;
#pragma unroll
  for (int bb = 0; bb < 4; ++bb) {
    const int b = bb * 2 + bsel;
    float4 v = hv[((size_t)b * NN + s) * 16 + c4];
    float* o = Out + ((size_t)b * NN + d) * HH + c4 * 4;
    atomicAdd(o,     w * v.x);
    atomicAdd(o + 1, w * v.y);
    atomicAdd(o + 2, w * v.z);
    atomicAdd(o + 3, w * v.w);
  }
}

// ---------------------------------------------------------------------------
// LayerNorm(eps=1e-3) over H + ReLU; wave per row. Writes f32 X (for the
// final projection) and the f16 mirror Xh (WMMA input for the next layer).
// ---------------------------------------------------------------------------
__global__ __launch_bounds__(256) void ln_relu(const float* __restrict__ Out,
                                               const float* __restrict__ g,
                                               const float* __restrict__ bta,
                                               float* __restrict__ X,
                                               _Float16* __restrict__ Xh) {
  const int row  = (blockIdx.x * 256 + threadIdx.x) >> 5;  // 80000 rows
  const int lane = threadIdx.x & 31;
  float2 v = ((const float2*)Out)[(size_t)row * 32 + lane];
  float s = v.x + v.y;
#pragma unroll
  for (int m = 16; m; m >>= 1) s += __shfl_xor(s, m, 32);
  const float mu = s * (1.0f / 64.0f);
  const float dx = v.x - mu, dy = v.y - mu;
  float q = dx * dx + dy * dy;
#pragma unroll
  for (int m = 16; m; m >>= 1) q += __shfl_xor(q, m, 32);
  const float rs = rsqrtf(q * (1.0f / 64.0f) + 1e-3f);
  float2 r;
  r.x = fmaxf(0.0f, fmaf(dx * rs, g[lane * 2],     bta[lane * 2]));
  r.y = fmaxf(0.0f, fmaf(dy * rs, g[lane * 2 + 1], bta[lane * 2 + 1]));
  ((float2*)X)[(size_t)row * 32 + lane] = r;
  v2h hv; hv[0] = (_Float16)r.x; hv[1] = (_Float16)r.y;
  ((v2h*)Xh)[(size_t)row * 32 + lane] = hv;
}

// ---------------------------------------------------------------------------
// Final projection: y[row] = x[row,:] . W_out + b_out ; wave per row.
// ---------------------------------------------------------------------------
__global__ __launch_bounds__(256) void out_proj(const float* __restrict__ X,
                                                const float* __restrict__ Wout,
                                                const float* __restrict__ bout,
                                                float* __restrict__ Y) {
  const int row  = (blockIdx.x * 256 + threadIdx.x) >> 5;
  const int lane = threadIdx.x & 31;
  float2 v = ((const float2*)X)[(size_t)row * 32 + lane];
  float s = v.x * Wout[lane * 2] + v.y * Wout[lane * 2 + 1];
#pragma unroll
  for (int m = 16; m; m >>= 1) s += __shfl_xor(s, m, 32);
  if (lane == 0) Y[row] = s + bout[0];
}

// ---------------------------------------------------------------------------
extern "C" void kernel_launch(void* const* d_in, const int* in_sizes, int n_in,
                              void* d_out, int out_size, void* d_ws, size_t ws_size,
                              hipStream_t stream) {
  const float* ctl   = (const float*)d_in[0];
  const float* tg    = (const float*)d_in[1];
  const float* dfp   = (const float*)d_in[2];
  const float* ew    = (const float*)d_in[3];
  const int*   cidx  = (const int*)  d_in[4];
  const int*   eidx  = (const int*)  d_in[5];
  const float* Wse   = (const float*)d_in[6];
  const float* bse   = (const float*)d_in[7];
  const float* cemb  = (const float*)d_in[8];
  const float* Wf1   = (const float*)d_in[9];
  const float* bf1   = (const float*)d_in[10];
  const float* Wf2   = (const float*)d_in[11];
  const float* bf2   = (const float*)d_in[12];
  const float* Wself = (const float*)d_in[13];
  const float* Wrel  = (const float*)d_in[14];
  const float* lng   = (const float*)d_in[15];
  const float* lnb   = (const float*)d_in[16];
  const float* Wout  = (const float*)d_in[17];
  const float* bout  = (const float*)d_in[18];

  const size_t ACT = (size_t)MROWS * HH;      // 5,120,000 elements
  float*    x    = (float*)d_ws;              // f32 activations (20.5 MB)
  float*    outb = x + ACT;                   // self-proj + messages (20.5 MB)
  float*    hbuf = outb + ACT;                // per-relation projection (20.5 MB)
  _Float16* xh   = (_Float16*)(hbuf + ACT);   // f16 activation mirror (10.25 MB)
  float*    film = (float*)(xh + ACT);        // [8][128]
  // total workspace use ~= 72 MB

  film_kernel<<<HB, 128, 0, stream>>>(dfp, Wf1, bf1, Wf2, bf2, film);
  init_x<<<(int)(ACT / 256), 256, 0, stream>>>(ctl, tg, Wse, bse, cemb, cidx, film, xh);

  const int gemm_blocks = STRIPS / (8 * SPW);       // 125
  const int scat_blocks = EE / 8;                   // 25000
  const int row_blocks  = MROWS / 8;                // 10000

  for (int l = 0; l < LL; ++l) {
    gemm64<<<gemm_blocks, 256, 0, stream>>>(xh, Wself + (size_t)l * HH * HH, outb);
    for (int r = 0; r < RR; ++r) {
      gemm64<<<gemm_blocks, 256, 0, stream>>>(
          xh, Wrel + ((size_t)l * RR + r) * HH * HH, hbuf);
      scatter_edges<<<scat_blocks, 256, 0, stream>>>(
          hbuf,
          eidx + (size_t)r * 2 * EE,        // src row
          eidx + (size_t)r * 2 * EE + EE,   // dst row
          ew + (size_t)r * EE,
          outb);
    }
    ln_relu<<<row_blocks, 256, 0, stream>>>(outb, lng + l * HH, lnb + l * HH, x, xh);
  }

  out_proj<<<row_blocks, 256, 0, stream>>>(x, Wout, bout, (float*)d_out);
}
